// CPRNN_11416023073124
// MI455X (gfx1250) — matrix operand order repaired and checked by reading, not compiled, not run on hardware
//
#include <hip/hip_runtime.h>
#include <hip/hip_bf16.h>

#define VOCAB 32000
#define HID   1024
#define EMB   512
#define RANK  64
#define BATCH 16
#define SEQ   256
#define ROWS  (SEQ * BATCH)   // 4096, row = s*16 + b

typedef __attribute__((ext_vector_type(16))) __bf16 bf16x16;
typedef __attribute__((ext_vector_type(8)))  __bf16 bf16x8;
typedef __attribute__((ext_vector_type(8)))  float  f32x8;
typedef __attribute__((ext_vector_type(4)))  unsigned int u32x4;
typedef __attribute__((ext_vector_type(8)))  int i32x8;
typedef __attribute__((ext_vector_type(4)))  int i32x4;

#if defined(__has_builtin)
#if __has_builtin(__builtin_amdgcn_tensor_load_to_lds) && \
    __has_builtin(__builtin_amdgcn_s_wait_tensorcnt)
#define HAVE_TDM 1
#endif
#if __has_builtin(__builtin_amdgcn_global_load_async_to_lds_b128) && \
    __has_builtin(__builtin_amdgcn_s_wait_asynccnt)
#define HAVE_ASYNC 1
#endif
#endif
#ifndef HAVE_TDM
#define HAVE_TDM 0
#endif
#ifndef HAVE_ASYNC
#define HAVE_ASYNC 0
#endif

union FragBF { bf16x16 v; bf16x8 h[2]; };

__device__ __forceinline__ __bf16 f2bf(float f) {
  unsigned u = __builtin_bit_cast(unsigned, f);
  u = (u + 0x7FFFu + ((u >> 16) & 1u)) >> 16;
  unsigned short s = (unsigned short)u;
  return __builtin_bit_cast(__bf16, s);
}

__device__ __forceinline__ f32x8 wmma_bf16(bf16x16 a, bf16x16 b, f32x8 c) {
  return __builtin_amdgcn_wmma_f32_16x16x32_bf16(false, a, false, b, (short)0, c,
                                                 false, false);
}

// A-fragment from a per-lane row pointer (row of M = lane&15 already applied).
__device__ __forceinline__ bf16x16 afrag_p(const __bf16* rowPtr, int k0, int lane) {
  int base = (lane >> 4) * 8;
  const __bf16* p = rowPtr + k0 + base;
  FragBF f;
  f.h[0] = *(const bf16x8*)p;
  f.h[1] = *(const bf16x8*)(p + 16);
  return f.v;
}

// A-fragment (16x32 bf16, MxK) from a row-major [*, ld] bf16 source.
__device__ __forceinline__ bf16x16 afrag(const __bf16* src, int ld, int k0, int lane) {
  return afrag_p(src + (size_t)(lane & 15) * ld, k0, lane);
}

// B-fragment (32x16 bf16, KxN) from an [N][K] row-major bf16 source.
__device__ __forceinline__ bf16x16 bfrag(const __bf16* srcNK, int ld, int n0, int k0,
                                         int lane) {
  int n = n0 + (lane & 15);
  int koff = k0 + ((lane >> 4) << 3);
  const __bf16* p = srcNK + (size_t)n * ld + koff;
  FragBF f;
  f.h[0] = *(const bf16x8*)p;
  f.h[1] = *(const bf16x8*)(p + 16);
  return f.v;
}

#if HAVE_ASYNC
typedef __attribute__((address_space(1))) i32x4 gas_i32x4;
typedef __attribute__((address_space(3))) i32x4 las_i32x4;
__device__ __forceinline__ void async_b128(const __bf16* g, __bf16* l) {
  __builtin_amdgcn_global_load_async_to_lds_b128((gas_i32x4*)g, (las_i32x4*)l, 0, 0);
}
#endif

#if HAVE_TDM
// 1-D contiguous bf16 DMA: global -> LDS via the Tensor Data Mover.
__device__ __forceinline__ void tdm_load_1d(unsigned lds_off, const void* gaddr,
                                            unsigned nElems) {
  unsigned long long ga = (unsigned long long)(uintptr_t)gaddr;
  u32x4 g0;
  g0[0] = 1u;                                              // count=1, user mode
  g0[1] = lds_off;                                         // lds_addr (bytes)
  g0[2] = (unsigned)(ga & 0xFFFFFFFFu);                    // global_addr[31:0]
  g0[3] = (unsigned)((ga >> 32) & 0x01FFFFFFu) | 0x80000000u; // addr[56:32] | type=2
  i32x8 g1;
  g1[0] = (int)(1u << 16);                 // data_size=1 (2 bytes), no flags
  g1[1] = (int)((nElems & 0xFFFFu) << 16); // tensor_dim0[15:0]
  g1[2] = (int)((nElems >> 16) | (1u << 16)); // tensor_dim0[31:16] | tensor_dim1=1
  g1[3] = (int)((nElems & 0xFFFFu) << 16); // tile_dim0
  g1[4] = 1;                               // tile_dim1=1, tile_dim2=0
  g1[5] = (int)nElems;                     // tensor_dim0_stride[31:0]
  g1[6] = 0;
  g1[7] = 0;
  i32x4 z4 = {0, 0, 0, 0};
#if __clang_major__ >= 23
  i32x8 z8 = {0, 0, 0, 0, 0, 0, 0, 0};
  __builtin_amdgcn_tensor_load_to_lds(g0, g1, z4, z4, z8, 0);
#else
  __builtin_amdgcn_tensor_load_to_lds(g0, g1, z4, z4, 0);
#endif
}
#endif

// ---------------- conversion kernels ----------------
__global__ void k_tconv(const float* __restrict__ src, __bf16* __restrict__ dst,
                        int R, int C) {
  long long idx = (long long)blockIdx.x * blockDim.x + threadIdx.x;
  if (idx >= (long long)R * C) return;
  int r = (int)(idx / C), c = (int)(idx % C);
  dst[(long long)c * R + r] = f2bf(src[idx]);
}

__global__ void k_conv(const float* __restrict__ src, __bf16* __restrict__ dst,
                       long long n) {
  long long idx = (long long)blockIdx.x * blockDim.x + threadIdx.x;
  if (idx < n) dst[idx] = f2bf(src[idx]);
}

// ---------------- xU = gather(emb) @ U + d  (bf16 out) ----------------
// grid (2, 256); block 256 (8 waves). wave -> 4 n-tiles; M-tile == s == blockIdx.y.
__global__ void k_xu(const int* __restrict__ inp, const __bf16* __restrict__ embbf,
                     const __bf16* __restrict__ Ut, const float* __restrict__ d,
                     __bf16* __restrict__ xU) {
  int lane = threadIdx.x & 31, wave = threadIdx.x >> 5;
  int g = blockIdx.x * 8 + wave;  // 0..15
  int s = blockIdx.y;
  int tok = inp[(lane & 15) * SEQ + s];
  const __bf16* rowBase = embbf + (size_t)tok * EMB;

  f32x8 acc[4] = {};
  bf16x16 a = afrag_p(rowBase, 0, lane);
  bf16x16 b[4];
#pragma unroll
  for (int i = 0; i < 4; ++i) b[i] = bfrag(Ut, EMB, (g * 4 + i) * 16, 0, lane);
  for (int ks = 1; ks <= EMB / 32; ++ks) {
    bf16x16 an;
    bf16x16 bn[4];
    if (ks < EMB / 32) {
      an = afrag_p(rowBase, ks * 32, lane);
#pragma unroll
      for (int i = 0; i < 4; ++i) bn[i] = bfrag(Ut, EMB, (g * 4 + i) * 16, ks * 32, lane);
    }
#pragma unroll
    for (int i = 0; i < 4; ++i) acc[i] = wmma_bf16(a, b[i], acc[i]);
    a = an;
#pragma unroll
    for (int i = 0; i < 4; ++i) b[i] = bn[i];
  }
  int n = lane & 15, m0 = (lane >> 4) * 8;
#pragma unroll
  for (int i = 0; i < 4; ++i) {
    int hcol = (g * 4 + i) * 16 + n;
    float dv = d[hcol];
#pragma unroll
    for (int j = 0; j < 8; ++j) {
      int bb = m0 + j;
      xU[((size_t)s * 16 + bb) * HID + hcol] = f2bf(acc[i][j] + dv);
    }
  }
}

// ---------------- xB = gather(emb) @ B ----------------
__global__ void k_xb(const int* __restrict__ inp, const __bf16* __restrict__ embbf,
                     const __bf16* __restrict__ Bt, float* __restrict__ xB) {
  int lane = threadIdx.x & 31, wave = threadIdx.x >> 5;
  int s = blockIdx.x;
  int tok = inp[(lane & 15) * SEQ + s];
  const __bf16* rowBase = embbf + (size_t)tok * EMB;

  f32x8 acc = {};
  for (int ks = 0; ks < EMB / 32; ++ks) {
    bf16x16 a = afrag_p(rowBase, ks * 32, lane);
    bf16x16 b = bfrag(Bt, EMB, wave * 16, ks * 32, lane);
    acc = wmma_bf16(a, b, acc);
  }
  int n = lane & 15, m0 = (lane >> 4) * 8;
#pragma unroll
  for (int j = 0; j < 8; ++j) {
    int bb = m0 + j;
    xB[((size_t)s * 16 + bb) * RANK + wave * 16 + n] = acc[j];
  }
}

// ---------------- sequential recurrence ----------------
struct RecurLDS {
  __bf16 xu[2][16 * HID];  // double-buffered xU step slices (TDM target)
  __bf16 hbuf[16 * HID];   // h in A-operand layout
  float  ApL[16 * RANK];   // h @ A
  __bf16 GL[16 * RANK];    // (h@A)*(x@B) as bf16 A-operand
};

__global__ void k_recur(const __bf16* __restrict__ xU, const float* __restrict__ xB,
                        const __bf16* __restrict__ At, const __bf16* __restrict__ Vt,
                        const __bf16* __restrict__ Cb, __bf16* __restrict__ hidden) {
  __shared__ RecurLDS L;
  int lane = threadIdx.x & 31, wave = threadIdx.x >> 5;
  int n = lane & 15, m0 = (lane >> 4) * 8;

  for (int i = threadIdx.x; i < 16 * HID; i += 1024) L.hbuf[i] = f2bf(0.0f);
#if HAVE_TDM
  if (wave == 0)
    tdm_load_1d((unsigned)(uintptr_t)&L.xu[0][0], xU, 16 * HID);
#endif
  __syncthreads();

  for (int s = 0; s < SEQ; ++s) {
#if HAVE_TDM
    if (wave == 0 && s + 1 < SEQ)
      tdm_load_1d((unsigned)(uintptr_t)&L.xu[(s + 1) & 1][0],
                  xU + (size_t)(s + 1) * 16 * HID, 16 * HID);
#endif
    // ---- hV = h @ V : wave owns n-tiles {2w, 2w+1} (register double-buffered)
    f32x8 hv0 = {}, hv1 = {};
    {
      bf16x16 a = afrag(L.hbuf, HID, 0, lane);
      bf16x16 b0 = bfrag(Vt, HID, (2 * wave) * 16, 0, lane);
      bf16x16 b1 = bfrag(Vt, HID, (2 * wave + 1) * 16, 0, lane);
      for (int ks = 1; ks <= HID / 32; ++ks) {
        bf16x16 an, b0n, b1n;
        if (ks < HID / 32) {
          an = afrag(L.hbuf, HID, ks * 32, lane);
          b0n = bfrag(Vt, HID, (2 * wave) * 16, ks * 32, lane);
          b1n = bfrag(Vt, HID, (2 * wave + 1) * 16, ks * 32, lane);
        }
        hv0 = wmma_bf16(a, b0, hv0);
        hv1 = wmma_bf16(a, b1, hv1);
        a = an; b0 = b0n; b1 = b1n;
      }
    }
    // ---- Ap = h @ A : waves 0..3 (uniform per-wave branch, EXEC stays full)
    if (wave < 4) {
      f32x8 ap = {};
      for (int ks = 0; ks < HID / 32; ++ks) {
        bf16x16 a = afrag(L.hbuf, HID, ks * 32, lane);
        bf16x16 b = bfrag(At, HID, wave * 16, ks * 32, lane);
        ap = wmma_bf16(a, b, ap);
      }
#pragma unroll
      for (int j = 0; j < 8; ++j) L.ApL[(m0 + j) * RANK + wave * 16 + n] = ap[j];
    }
#if HAVE_TDM
    if (wave == 0) {
      if (s == SEQ - 1) __builtin_amdgcn_s_wait_tensorcnt(0);
      else              __builtin_amdgcn_s_wait_tensorcnt(1);
    }
#endif
    __syncthreads();  // Ap ready; xU slice for step s landed; h reads complete

    // ---- G = Ap * Bp (elementwise), one element per thread
    {
      int bb = threadIdx.x >> 6, r = threadIdx.x & 63;
      float g = L.ApL[bb * RANK + r] * xB[((size_t)s * 16 + bb) * RANK + r];
      L.GL[bb * RANK + r] = f2bf(g);
    }
    __syncthreads();  // G ready

    // ---- cp = G @ C^T  (B-operand is C's native [h][r] layout)
    f32x8 cp0 = {}, cp1 = {};
#pragma unroll
    for (int ks = 0; ks < RANK / 32; ++ks) {
      bf16x16 a = afrag(L.GL, RANK, ks * 32, lane);
      bf16x16 b0 = bfrag(Cb, RANK, (2 * wave) * 16, ks * 32, lane);
      bf16x16 b1 = bfrag(Cb, RANK, (2 * wave + 1) * 16, ks * 32, lane);
      cp0 = wmma_bf16(a, b0, cp0);
      cp1 = wmma_bf16(a, b1, cp1);
    }

    // ---- h_new = tanh(cp + hV + xU); write LDS h and global hidden (bf16)
    f32x8 acc[2];
    acc[0] = hv0 + cp0;
    acc[1] = hv1 + cp1;
#pragma unroll
    for (int t = 0; t < 2; ++t) {
      int nt = 2 * wave + t;
#pragma unroll
      for (int j = 0; j < 8; ++j) {
        int bb = m0 + j, hcol = nt * 16 + n;
        size_t row = (size_t)s * 16 + bb;
#if HAVE_TDM
        float xv = (float)L.xu[s & 1][bb * HID + hcol];
#else
        float xv = (float)xU[row * HID + hcol];
#endif
        float v = acc[t][j] + xv;
        __bf16 hb = f2bf(tanhf(v));
        L.hbuf[bb * HID + hcol] = hb;
        hidden[row * HID + hcol] = hb;
      }
    }
    __syncthreads();  // h_new visible; GL/xu-buffer reads done before reuse
  }
}

// ---------------- decoder: out = hidden @ W_dec + b_dec ----------------
// grid (250, 64); block 128 (4 waves). Each wave: 4 M-tiles x 2 N-tiles.
// Async-to-LDS double-buffered k-chunks: DMA of chunk ks+1 overlaps WMMAs of ks.
#if HAVE_ASYNC
struct DecWaveLDS {
  __bf16 Ab[2][64 * 32];   // 4KB per buffer: A chunk [row][k], ld=32
  __bf16 Bb[2][32 * 32];   // 2KB per buffer: B chunk [col][k], ld=32
};
#endif

__global__ void k_dec(const __bf16* __restrict__ hid, const __bf16* __restrict__ Wdt,
                      const float* __restrict__ bias, float* __restrict__ out) {
  int lane = threadIdx.x & 31, wave = threadIdx.x >> 5;
  int rowBase = blockIdx.y * 64;         // 4 m-tiles = 64 rows
  int nt0 = blockIdx.x * 8 + wave * 2;   // 2 n-tiles per wave

  f32x8 acc[4][2] = {};

#if HAVE_ASYNC
  __shared__ DecWaveLDS LD[4];
  DecWaveLDS& W = LD[wave];

  // Issue one k-chunk (32 k) of A (64 rows, wave-private) + B (32 cols) via ASYNC DMA.
  auto issueChunk = [&](int ks, int buf) {
#pragma unroll
    for (int i = 0; i < 8; ++i) {          // A: 4KB = 8 x b128 per lane
      int eo = (lane + i * 32) * 8;
      int row = eo >> 5, kk = eo & 31;
      async_b128(hid + (size_t)(rowBase + row) * HID + ks * 32 + kk, &W.Ab[buf][eo]);
    }
#pragma unroll
    for (int i = 0; i < 4; ++i) {          // B: 2KB = 4 x b128 per lane
      int eo = (lane + i * 32) * 8;
      int col = eo >> 5, kk = eo & 31;
      async_b128(Wdt + (size_t)(nt0 * 16 + col) * HID + ks * 32 + kk, &W.Bb[buf][eo]);
    }
  };

  issueChunk(0, 0);
  for (int ks = 0; ks < HID / 32; ++ks) {
    int buf = ks & 1;
    if (ks + 1 < HID / 32) {
      issueChunk(ks + 1, buf ^ 1);
      __builtin_amdgcn_s_wait_asynccnt(12);  // chunk ks landed; ks+1 in flight
    } else {
      __builtin_amdgcn_s_wait_asynccnt(0);
    }
    bf16x16 b0 = bfrag(W.Bb[buf], 32, 0, 0, lane);
    bf16x16 b1 = bfrag(W.Bb[buf], 32, 16, 0, lane);
#pragma unroll
    for (int mi = 0; mi < 4; ++mi) {
      bf16x16 a = afrag(W.Ab[buf] + mi * 16 * 32, 32, 0, lane);
      acc[mi][0] = wmma_bf16(a, b0, acc[mi][0]);
      acc[mi][1] = wmma_bf16(a, b1, acc[mi][1]);
    }
  }
#else
  const __bf16* aRow[4];
#pragma unroll
  for (int mi = 0; mi < 4; ++mi)
    aRow[mi] = hid + ((size_t)(rowBase + mi * 16) + (lane & 15)) * HID;

  bf16x16 a[4], b[2];
#pragma unroll
  for (int mi = 0; mi < 4; ++mi) a[mi] = afrag_p(aRow[mi], 0, lane);
  b[0] = bfrag(Wdt, HID, nt0 * 16, 0, lane);
  b[1] = bfrag(Wdt, HID, (nt0 + 1) * 16, 0, lane);

  for (int ks = 1; ks <= HID / 32; ++ks) {
    bf16x16 an[4], bn[2];
    if (ks < HID / 32) {
#pragma unroll
      for (int mi = 0; mi < 4; ++mi) an[mi] = afrag_p(aRow[mi], ks * 32, lane);
      bn[0] = bfrag(Wdt, HID, nt0 * 16, ks * 32, lane);
      bn[1] = bfrag(Wdt, HID, (nt0 + 1) * 16, ks * 32, lane);
    }
#pragma unroll
    for (int mi = 0; mi < 4; ++mi) {
      acc[mi][0] = wmma_bf16(a[mi], b[0], acc[mi][0]);
      acc[mi][1] = wmma_bf16(a[mi], b[1], acc[mi][1]);
    }
#pragma unroll
    for (int mi = 0; mi < 4; ++mi) a[mi] = an[mi];
    b[0] = bn[0]; b[1] = bn[1];
  }
#endif

  int n = lane & 15, m0 = (lane >> 4) * 8;
#pragma unroll
  for (int mi = 0; mi < 4; ++mi) {
    int s = blockIdx.y * 4 + mi;
#pragma unroll
    for (int i = 0; i < 2; ++i) {
      int v = (nt0 + i) * 16 + n;
      float bv = bias[v];
#pragma unroll
      for (int j = 0; j < 8; ++j) {
        int bb = m0 + j;
        out[(size_t)bb * SEQ * VOCAB + (size_t)s * VOCAB + v] = acc[mi][i][j] + bv;
      }
    }
  }
}

extern "C" void kernel_launch(void* const* d_in, const int* in_sizes, int n_in,
                              void* d_out, int out_size, void* d_ws, size_t ws_size,
                              hipStream_t stream) {
  const int*   inp  = (const int*)d_in[0];
  const float* emb  = (const float*)d_in[1];
  const float* A    = (const float*)d_in[2];
  const float* B    = (const float*)d_in[3];
  const float* C    = (const float*)d_in[4];
  const float* U    = (const float*)d_in[5];
  const float* V    = (const float*)d_in[6];
  const float* d    = (const float*)d_in[7];
  const float* Wd   = (const float*)d_in[8];
  const float* bdec = (const float*)d_in[9];
  float* out = (float*)d_out;

  char* ws = (char*)d_ws;
  size_t o = 0;
  __bf16* Wdt   = (__bf16*)(ws + o); o += (size_t)VOCAB * HID * 2;  // 65.5 MB
  __bf16* Vt    = (__bf16*)(ws + o); o += (size_t)HID * HID * 2;    //  2 MB
  __bf16* Ut    = (__bf16*)(ws + o); o += (size_t)HID * EMB * 2;
  __bf16* At    = (__bf16*)(ws + o); o += (size_t)RANK * HID * 2;
  __bf16* Bt    = (__bf16*)(ws + o); o += (size_t)RANK * EMB * 2;
  __bf16* Cb    = (__bf16*)(ws + o); o += (size_t)HID * RANK * 2;
  __bf16* embbf = (__bf16*)(ws + o); o += (size_t)VOCAB * EMB * 2;  // 32.8 MB
  __bf16* xU    = (__bf16*)(ws + o); o += (size_t)ROWS * HID * 2;   //  8.4 MB
  float*  xB    = (float*)(ws + o);  o += (size_t)ROWS * RANK * 4;
  __bf16* hid   = (__bf16*)(ws + o); o += (size_t)ROWS * HID * 2;   //  8.4 MB

  auto blocks = [](long long nn) { return (unsigned)((nn + 255) / 256); };

  // weight conversion / transposition
  k_tconv<<<blocks((long long)HID * RANK), 256, 0, stream>>>(A, At, HID, RANK);
  k_tconv<<<blocks((long long)EMB * RANK), 256, 0, stream>>>(B, Bt, EMB, RANK);
  k_tconv<<<blocks((long long)EMB * HID), 256, 0, stream>>>(U, Ut, EMB, HID);
  k_tconv<<<blocks((long long)HID * HID), 256, 0, stream>>>(V, Vt, HID, HID);
  k_tconv<<<blocks((long long)HID * VOCAB), 256, 0, stream>>>(Wd, Wdt, HID, VOCAB);
  k_conv<<<blocks((long long)HID * RANK), 256, 0, stream>>>(C, Cb, (long long)HID * RANK);
  k_conv<<<blocks((long long)VOCAB * EMB), 256, 0, stream>>>(emb, embbf,
                                                             (long long)VOCAB * EMB);

  // token-parallel precompute
  k_xu<<<dim3(2, SEQ), 256, 0, stream>>>(inp, embbf, Ut, d, xU);
  k_xb<<<dim3(SEQ), 128, 0, stream>>>(inp, embbf, Bt, xB);

  // sequential recurrence (one persistent workgroup, TDM-fed xU)
  k_recur<<<1, 1024, 0, stream>>>(xU, xB, At, Vt, Cb, hid);

  // decoder GEMM (async-to-LDS double-buffered)
  k_dec<<<dim3(VOCAB / 128, ROWS / 64), 128, 0, stream>>>(hid, Wdt, bdec, out);
}
